// MultiStepTransformer_47708496724827
// MI455X (gfx1250) — compile-verified, matrix operand order checked by reading, not run on hardware
//
#include <hip/hip_runtime.h>
#include <hip/hip_bf16.h>
#include <math.h>

// ---------------------------------------------------------------------------
// CDNA5 (gfx1250) WMMA types
// ---------------------------------------------------------------------------
typedef __attribute__((ext_vector_type(16))) __bf16 v16bf;
typedef __attribute__((ext_vector_type(8)))  __bf16 v8bf;
typedef __attribute__((ext_vector_type(4)))  __bf16 v4bf;
typedef __attribute__((ext_vector_type(8)))  float  v8f;

#define V8ZERO ((v8f){0.f,0.f,0.f,0.f,0.f,0.f,0.f,0.f})

#define WMMA_BF16(a, b, c) \
  __builtin_amdgcn_wmma_f32_16x16x32_bf16(false, (a), false, (b), (short)0, (c), false, false)

// CDNA5 16-bit fragment layout (ISA 7.12.2): for one lane the 16 elements of a
// 16x32 A/B fragment are K = {8*laneHi..+7} and {16+8*laneHi..+7} of one row.
// With K-contiguous LDS rows that is two aligned 16-byte chunks -> 2x ds_load_b128.
__device__ __forceinline__ v16bf load_frag(const __bf16* rowp, int laneHi) {
  const v8bf* rp = (const v8bf*)rowp;
  v8bf lo = rp[laneHi];        // halfwords 8*laneHi    .. +7
  v8bf hi = rp[2 + laneHi];    // halfwords 16+8*laneHi .. +7
  return __builtin_shufflevector(lo, hi, 0,1,2,3,4,5,6,7,8,9,10,11,12,13,14,15);
}

__device__ __forceinline__ v4bf cvt4(float4 v) {
  return (v4bf){(__bf16)v.x, (__bf16)v.y, (__bf16)v.z, (__bf16)v.w};
}

__device__ __forceinline__ float redmax16(float v) {
  v = fmaxf(v, __shfl_xor(v, 1, 32));
  v = fmaxf(v, __shfl_xor(v, 2, 32));
  v = fmaxf(v, __shfl_xor(v, 4, 32));
  v = fmaxf(v, __shfl_xor(v, 8, 32));
  return v;
}
__device__ __forceinline__ float redsum16(float v) {
  v += __shfl_xor(v, 1, 32);
  v += __shfl_xor(v, 2, 32);
  v += __shfl_xor(v, 4, 32);
  v += __shfl_xor(v, 8, 32);
  return v;
}

// ---------------------------------------------------------------------------
// GEMM: C[M,N] = act(A[M,K] @ W[N,K]^T + bias[N])   (torch Linear convention)
// A, W f32 in global, converted to bf16 while staging (float4 loads, b64 stores).
// Block: 4 waves / 128 threads; tile 128x64; each wave computes 32x64 so every
// B fragment is reused by two WMMAs (DS loads per WMMA: 24 b128 / 16 wmma).
// ---------------------------------------------------------------------------
#define GBM 128
#define GBN 64
#define GBK 64
#define GSTR 72   // row stride in halfwords: 144B, 16B-aligned, odd*16B -> bank rotate

__global__ __launch_bounds__(128) void gemm_wmma(
    const float* __restrict__ A, const float* __restrict__ W,
    const float* __restrict__ bias, float* __restrict__ C,
    int M, int N, int K, int relu)
{
  __shared__ __align__(16) __bf16 As[GBM * GSTR];
  __shared__ __align__(16) __bf16 Ws[GBN * GSTR];

  const int tid    = threadIdx.x;
  const int lane   = tid & 31;
  const int wave   = tid >> 5;
  const int lan15  = lane & 15;
  const int laneHi = lane >> 4;
  const int bm = blockIdx.y * GBM;
  const int bn = blockIdx.x * GBN;

  v8f acc0[4] = {V8ZERO, V8ZERO, V8ZERO, V8ZERO};   // rows wave*32 + 0..15
  v8f acc1[4] = {V8ZERO, V8ZERO, V8ZERO, V8ZERO};   // rows wave*32 + 16..31

  for (int k0 = 0; k0 < K; k0 += GBK) {
    // prefetch next K tile (global_prefetch_b8)
    if (k0 + GBK < K) {
      if (bm + tid < M) __builtin_prefetch(&A[(size_t)(bm + tid) * K + k0 + GBK], 0, 1);
      if (tid < GBN && bn + tid < N)
        __builtin_prefetch(&W[(size_t)(bn + tid) * K + k0 + GBK], 0, 1);
    }
    // stage A tile (128x64): float4 global loads, packed 4xbf16 LDS stores
    for (int i = tid; i < GBM * GBK / 4; i += 128) {
      int r = i >> 4, c4 = (i & 15) * 4;
      int gr = bm + r;
      float4 v = (gr < M) ? *(const float4*)&A[(size_t)gr * K + k0 + c4]
                          : make_float4(0.f, 0.f, 0.f, 0.f);
      *(v4bf*)&As[r * GSTR + c4] = cvt4(v);
    }
    // stage W tile (64x64) untransposed: Ws[n][k]
    for (int i = tid; i < GBN * GBK / 4; i += 128) {
      int n = i >> 4, c4 = (i & 15) * 4;
      int gn = bn + n;
      float4 v = (gn < N) ? *(const float4*)&W[(size_t)gn * K + k0 + c4]
                          : make_float4(0.f, 0.f, 0.f, 0.f);
      *(v4bf*)&Ws[n * GSTR + c4] = cvt4(v);
    }
    __syncthreads();

    const int arow0 = wave * 32 + lan15;
    const int arow1 = arow0 + 16;
    v16bf a00 = load_frag(&As[arow0 * GSTR],      laneHi);
    v16bf a01 = load_frag(&As[arow0 * GSTR + 32], laneHi);
    v16bf a10 = load_frag(&As[arow1 * GSTR],      laneHi);
    v16bf a11 = load_frag(&As[arow1 * GSTR + 32], laneHi);
#pragma unroll
    for (int ct = 0; ct < 4; ++ct) {
      const int col = ct * 16 + lan15;
      v16bf b0 = load_frag(&Ws[col * GSTR],      laneHi);
      v16bf b1 = load_frag(&Ws[col * GSTR + 32], laneHi);
      acc0[ct] = WMMA_BF16(a00, b0, acc0[ct]);
      acc0[ct] = WMMA_BF16(a01, b1, acc0[ct]);
      acc1[ct] = WMMA_BF16(a10, b0, acc1[ct]);
      acc1[ct] = WMMA_BF16(a11, b1, acc1[ct]);
    }
    __syncthreads();
  }

  // epilogue: bias (+ optional ReLU), f32 store
#pragma unroll
  for (int ct = 0; ct < 4; ++ct) {
    int gcol = bn + ct * 16 + lan15;
    if (gcol >= N) continue;
    float bv = bias ? bias[gcol] : 0.f;
#pragma unroll
    for (int r = 0; r < 8; ++r) {
      int row0 = bm + wave * 32 + r + laneHi * 8;
      if (row0 < M) {
        float v = acc0[ct][r] + bv;
        if (relu) v = fmaxf(v, 0.f);
        C[(size_t)row0 * N + gcol] = v;
      }
      int row1 = row0 + 16;
      if (row1 < M) {
        float v = acc1[ct][r] + bv;
        if (relu) v = fmaxf(v, 0.f);
        C[(size_t)row1 * N + gcol] = v;
      }
    }
  }
}

// ---------------------------------------------------------------------------
// Flash attention. 4 waves/block; each wave owns a 16-row q tile of the same
// (b, head); the block shares K/V staging. Keys streamed in blocks of 32.
// Q/K/V f32 [rows,1024], head hd at cols hd*64..+63. All matmuls bf16 WMMA.
// ---------------------------------------------------------------------------
#define ADIM 1024
#define ADK  64
#define FQW  4          // waves per block
#define QSTR 72
#define PSTR 40

__global__ __launch_bounds__(32 * FQW) void flash_wmma(
    const float* __restrict__ Q, const float* __restrict__ Kg,
    const float* __restrict__ Vg, float* __restrict__ O,
    int Tq, int Tk)
{
  __shared__ __align__(16) __bf16 Qs[16 * FQW * QSTR];
  __shared__ __align__(16) __bf16 Ks[32 * QSTR];
  __shared__ __align__(16) __bf16 Vst[ADK * PSTR];       // [dk][key]
  __shared__ __align__(16) __bf16 Ps[FQW * 16 * PSTR];   // per-wave P tiles

  const int tid    = threadIdx.x;
  const int lane   = tid & 31;
  const int wave   = tid >> 5;
  const int lan15  = lane & 15;
  const int laneHi = lane >> 4;
  const int b  = blockIdx.x >> 4;   // H = 16
  const int hd = blockIdx.x & 15;
  const int qblk = blockIdx.y * (16 * FQW);
  const int q0 = qblk + wave * 16;

  const size_t qbase = ((size_t)b * Tq) * ADIM + (size_t)hd * ADK;
  const size_t kbase = ((size_t)b * Tk) * ADIM + (size_t)hd * ADK;

  // stage Q tiles (float4 loads); fold in 1/sqrt(64) = 0.125 (exact in bf16)
  for (int i = tid; i < 16 * FQW * ADK / 4; i += 32 * FQW) {
    int r = i >> 4, c4 = (i & 15) * 4;
    float4 v = make_float4(0.f, 0.f, 0.f, 0.f);
    if (qblk + r < Tq) v = *(const float4*)&Q[qbase + (size_t)(qblk + r) * ADIM + c4];
    v.x *= 0.125f; v.y *= 0.125f; v.z *= 0.125f; v.w *= 0.125f;
    *(v4bf*)&Qs[r * QSTR + c4] = cvt4(v);
  }
  __syncthreads();

  const int qrow = wave * 16 + lan15;
  v16bf aq0 = load_frag(&Qs[qrow * QSTR],      laneHi);
  v16bf aq1 = load_frag(&Qs[qrow * QSTR + 32], laneHi);

  float m[8], l[8];
#pragma unroll
  for (int r = 0; r < 8; ++r) { m[r] = -1e30f; l[r] = 0.f; }
  v8f accO[4] = {V8ZERO, V8ZERO, V8ZERO, V8ZERO};

  const int nkb = (Tk + 31) / 32;
  for (int kb = 0; kb < nkb; ++kb) {
    const int kstart = kb * 32;
    // stage K (row-major, packed stores) and V (transposed) blocks
    for (int i = tid; i < 32 * ADK / 4; i += 32 * FQW) {
      int r = i >> 4, c4 = (i & 15) * 4;
      float4 kv = make_float4(0.f, 0.f, 0.f, 0.f);
      float4 vv = make_float4(0.f, 0.f, 0.f, 0.f);
      if (kstart + r < Tk) {
        size_t off = kbase + (size_t)(kstart + r) * ADIM + c4;
        kv = *(const float4*)&Kg[off];
        vv = *(const float4*)&Vg[off];
      }
      *(v4bf*)&Ks[r * QSTR + c4] = cvt4(kv);
      Vst[(c4 + 0) * PSTR + r] = (__bf16)vv.x;
      Vst[(c4 + 1) * PSTR + r] = (__bf16)vv.y;
      Vst[(c4 + 2) * PSTR + r] = (__bf16)vv.z;
      Vst[(c4 + 3) * PSTR + r] = (__bf16)vv.w;
    }
    __syncthreads();

    // S tiles: s0 = keys [kstart..+15], s1 = [kstart+16..+31]
    v8f s0 = V8ZERO, s1 = V8ZERO;
    {
      v16bf b0 = load_frag(&Ks[lan15 * QSTR],      laneHi);
      v16bf b1 = load_frag(&Ks[lan15 * QSTR + 32], laneHi);
      s0 = WMMA_BF16(aq0, b0, s0);
      s0 = WMMA_BF16(aq1, b1, s0);
    }
    {
      v16bf b0 = load_frag(&Ks[(16 + lan15) * QSTR],      laneHi);
      v16bf b1 = load_frag(&Ks[(16 + lan15) * QSTR + 32], laneHi);
      s1 = WMMA_BF16(aq0, b0, s1);
      s1 = WMMA_BF16(aq1, b1, s1);
    }
    // mask padded keys
    if (kstart + lan15 >= Tk) {
#pragma unroll
      for (int r = 0; r < 8; ++r) s0[r] = -1e30f;
    }
    if (kstart + 16 + lan15 >= Tk) {
#pragma unroll
      for (int r = 0; r < 8; ++r) s1[r] = -1e30f;
    }

    // online softmax; write P (bf16) to this wave's LDS tile for transpose
    __bf16* Pw = &Ps[wave * 16 * PSTR];
#pragma unroll
    for (int r = 0; r < 8; ++r) {
      float t  = fmaxf(s0[r], s1[r]);
      float rm = redmax16(t);
      float mn = fmaxf(m[r], rm);
      float sc = __expf(m[r] - mn);
      float p0 = __expf(s0[r] - mn);
      float p1 = __expf(s1[r] - mn);
      float rs = redsum16(p0 + p1);
      l[r] = l[r] * sc + rs;
      m[r] = mn;
#pragma unroll
      for (int ct = 0; ct < 4; ++ct) accO[ct][r] *= sc;
      int prow = r + laneHi * 8;
      Pw[prow * PSTR + lan15]      = (__bf16)p0;
      Pw[prow * PSTR + 16 + lan15] = (__bf16)p1;
    }
    __syncthreads();

    // O += P(16x32) x V(32x64); B[key][dk] frag: row = dk col, K = key.
    v16bf ap = load_frag(&Pw[lan15 * PSTR], laneHi);
#pragma unroll
    for (int ct = 0; ct < 4; ++ct) {
      const int col = ct * 16 + lan15;
      v16bf bv = load_frag(&Vst[col * PSTR], laneHi);
      accO[ct] = WMMA_BF16(ap, bv, accO[ct]);
    }
    __syncthreads();  // before next K/V staging overwrite
  }

  // write normalized output
#pragma unroll
  for (int r = 0; r < 8; ++r) {
    int grow = q0 + r + laneHi * 8;
    if (grow >= Tq) continue;
    float inv = 1.f / l[r];
#pragma unroll
    for (int ct = 0; ct < 4; ++ct) {
      O[qbase + (size_t)grow * ADIM + ct * 16 + lan15] = accO[ct][r] * inv;
    }
  }
}

// ---------------------------------------------------------------------------
// Small helper kernels (f32)
// ---------------------------------------------------------------------------
__global__ void pe_init(float* __restrict__ pe, int T, int D) {
  int idx = blockIdx.x * blockDim.x + threadIdx.x;
  if (idx >= T * D) return;
  int t = idx / D, c = idx % D;
  int i2 = c & ~1;
  float div = expf(-(float)i2 * (logf(10000.f) / (float)D));
  float a = (float)t * div;
  pe[idx] = (c & 1) ? cosf(a) : sinf(a);
}

__global__ void dec_init(float* __restrict__ dec) {  // zero dec[b][0][:]
  int idx = blockIdx.x * blockDim.x + threadIdx.x;
  if (idx >= 4 * 1024) return;
  int b = idx >> 10, c = idx & 1023;
  dec[((size_t)b * 25) * 1024 + c] = 0.f;
}

// dst[(b*Tn+t)*D+c] = src[(b*srcTstride+t)*D+c] + pe[t*D+c]
__global__ void add_pe(const float* __restrict__ src, const float* __restrict__ pe,
                       float* __restrict__ dst, int Bn, int Tn, int srcTstride, int D)
{
  int idx = blockIdx.x * blockDim.x + threadIdx.x;
  if (idx >= Bn * Tn * D) return;
  int c = idx % D;
  int t = (idx / D) % Tn;
  int b = idx / (D * Tn);
  dst[((size_t)b * Tn + t) * D + c] =
      src[((size_t)b * srcTstride + t) * D + c] + pe[(size_t)t * D + c];
}

// Out[row] = LN(A[row] + Bres[row]) * g + beta    (one block per row, D=1024)
__global__ __launch_bounds__(256) void ln_kernel(
    const float* __restrict__ A, const float* __restrict__ Bres,
    const float* __restrict__ g, const float* __restrict__ beta,
    float* __restrict__ Out, int D)
{
  const size_t row = blockIdx.x;
  __shared__ float redS[8], redQ[8];
  float s = 0.f, s2 = 0.f;
  for (int c = threadIdx.x; c < D; c += 256) {
    float v = A[row * D + c] + Bres[row * D + c];
    s += v; s2 += v * v;
  }
  for (int msk = 1; msk < 32; msk <<= 1) {
    s += __shfl_xor(s, msk, 32);
    s2 += __shfl_xor(s2, msk, 32);
  }
  int wave = threadIdx.x >> 5, lane = threadIdx.x & 31;
  if (lane == 0) { redS[wave] = s; redQ[wave] = s2; }
  __syncthreads();
  s = 0.f; s2 = 0.f;
  for (int i = 0; i < 8; ++i) { s += redS[i]; s2 += redQ[i]; }
  float mu = s / (float)D;
  float var = s2 / (float)D - mu * mu;
  float rs = rsqrtf(var + 1e-5f);
  for (int c = threadIdx.x; c < D; c += 256) {
    float v = A[row * D + c] + Bres[row * D + c];
    Out[row * D + c] = (v - mu) * rs * g[c] + beta[c];
  }
}

// step = dot(d[b, Tq-1, :], out_W) + out_b; preds[b, s] = step;
// dec[b, s+1, :] = step * temb_W + temb_b
__global__ __launch_bounds__(256) void step_out(
    const float* __restrict__ dcur, int Tq,
    const float* __restrict__ outW, const float* __restrict__ outb,
    const float* __restrict__ tembW, const float* __restrict__ tembb,
    float* __restrict__ preds, float* __restrict__ dec, int s)
{
  const int b = blockIdx.x;
  const float* row = dcur + ((size_t)b * Tq + (Tq - 1)) * 1024;
  __shared__ float red[8];
  __shared__ float stepv;
  float acc = 0.f;
  for (int c = threadIdx.x; c < 1024; c += 256) acc += row[c] * outW[c];
  for (int msk = 1; msk < 32; msk <<= 1) acc += __shfl_xor(acc, msk, 32);
  int wave = threadIdx.x >> 5, lane = threadIdx.x & 31;
  if (lane == 0) red[wave] = acc;
  __syncthreads();
  if (threadIdx.x == 0) {
    float t = outb[0];
    for (int i = 0; i < 8; ++i) t += red[i];
    stepv = t;
    preds[b * 24 + s] = t;
  }
  __syncthreads();
  float sv = stepv;
  for (int c = threadIdx.x; c < 1024; c += 256)
    dec[((size_t)b * 25 + (s + 1)) * 1024 + c] = sv * tembW[c] + tembb[c];
}

// ---------------------------------------------------------------------------
// Host orchestration
// ---------------------------------------------------------------------------
static inline void launch_gemm(const float* A, const float* W, const float* bias,
                               float* C, int M, int N, int K, int relu,
                               hipStream_t stream)
{
  dim3 grid((N + GBN - 1) / GBN, (M + GBM - 1) / GBM);
  gemm_wmma<<<grid, 128, 0, stream>>>(A, W, bias, C, M, N, K, relu);
}

extern "C" void kernel_launch(void* const* d_in, const int* in_sizes, int n_in,
                              void* d_out, int out_size, void* d_ws, size_t ws_size,
                              hipStream_t stream)
{
  (void)in_sizes; (void)n_in; (void)out_size; (void)ws_size;

  const int Bc = 4, T = 1024, D = 1024, Hh = 16, Lc = 4, DFF = 4096, NS = 24;

  const float* x     = (const float*)d_in[0];
  const float* inW   = (const float*)d_in[1];
  const float* inb   = (const float*)d_in[2];
  const float* encWq = (const float*)d_in[3];
  const float* encbq = (const float*)d_in[4];
  const float* encWk = (const float*)d_in[5];
  const float* encbk = (const float*)d_in[6];
  const float* encWv = (const float*)d_in[7];
  const float* encbv = (const float*)d_in[8];
  const float* encWo = (const float*)d_in[9];
  const float* encbo = (const float*)d_in[10];
  const float* decWq = (const float*)d_in[11];
  const float* decbq = (const float*)d_in[12];
  const float* decWk = (const float*)d_in[13];
  const float* decbk = (const float*)d_in[14];
  const float* decWv = (const float*)d_in[15];
  const float* decbv = (const float*)d_in[16];
  const float* decWo = (const float*)d_in[17];
  const float* decbo = (const float*)d_in[18];
  const float* ffeW1 = (const float*)d_in[19];
  const float* ffeb1 = (const float*)d_in[20];
  const float* ffeW2 = (const float*)d_in[21];
  const float* ffeb2 = (const float*)d_in[22];
  const float* ffdW1 = (const float*)d_in[23];
  const float* ffdb1 = (const float*)d_in[24];
  const float* ffdW2 = (const float*)d_in[25];
  const float* ffdb2 = (const float*)d_in[26];
  const float* neg   = (const float*)d_in[27];
  const float* neb   = (const float*)d_in[28];
  const float* ndg   = (const float*)d_in[29];
  const float* ndb   = (const float*)d_in[30];
  const float* outW  = (const float*)d_in[31];
  const float* outb  = (const float*)d_in[32];
  const float* tembW = (const float*)d_in[33];
  const float* tembb = (const float*)d_in[34];

  // workspace carve (f32)
  float* p = (float*)d_ws;
  auto take = [&](size_t n) { float* r = p; p += n; return r; };
  const size_t MT = (size_t)Bc * T;           // 4096 rows
  float* pe     = take((size_t)T * D);
  float* h      = take(MT * D);
  float* qf     = take(MT * D);
  float* kf     = take(MT * D);
  float* vf     = take(MT * D);
  float* attnf  = take(MT * D);
  float* tmp1   = take(MT * D);
  float* ffn1   = take(MT * DFF);
  float* ffn2   = take(MT * D);
  float* crossK = take((size_t)Lc * MT * D);
  float* crossV = take((size_t)Lc * MT * D);
  float* dec    = take((size_t)Bc * 25 * D);
  float* dxa    = take((size_t)96 * D);
  float* dq     = take((size_t)96 * D);
  float* dk     = take((size_t)96 * D);
  float* dv     = take((size_t)96 * D);
  float* dattn  = take((size_t)96 * D);
  float* dt1    = take((size_t)96 * D);
  float* dt2    = take((size_t)96 * D);
  float* dffn1  = take((size_t)96 * DFF);
  float* dffn2  = take((size_t)96 * D);
  float* preds  = (float*)d_out;

  // --- setup ---
  pe_init<<<dim3((T * D + 255) / 256), 256, 0, stream>>>(pe, T, D);
  dec_init<<<dim3((Bc * D + 255) / 256), 256, 0, stream>>>(dec);

  // --- input projection + positional encoding ---
  launch_gemm(x, inW, inb, h, (int)MT, D, 64, 0, stream);
  add_pe<<<dim3((int)((MT * D + 255) / 256)), 256, 0, stream>>>(h, pe, h, Bc, T, T, D);

  // --- encoder: h = LN(mha(h,h) + FFN(mha(h,h))) per layer ---
  for (int l = 0; l < Lc; ++l) {
    const float* Wq = encWq + (size_t)l * D * D; const float* bq = encbq + (size_t)l * D;
    const float* Wk = encWk + (size_t)l * D * D; const float* bk = encbk + (size_t)l * D;
    const float* Wv = encWv + (size_t)l * D * D; const float* bv = encbv + (size_t)l * D;
    const float* Wo = encWo + (size_t)l * D * D; const float* bo = encbo + (size_t)l * D;

    launch_gemm(h, Wq, bq, qf, (int)MT, D, D, 0, stream);
    launch_gemm(h, Wk, bk, kf, (int)MT, D, D, 0, stream);
    launch_gemm(h, Wv, bv, vf, (int)MT, D, D, 0, stream);
    flash_wmma<<<dim3(Bc * Hh, T / (16 * FQW)), 32 * FQW, 0, stream>>>(
        qf, kf, vf, attnf, T, T);
    launch_gemm(attnf, Wo, bo, tmp1, (int)MT, D, D, 0, stream);   // no residual in ref
    launch_gemm(tmp1, ffeW1, ffeb1, ffn1, (int)MT, DFF, D, 1, stream);
    launch_gemm(ffn1, ffeW2, ffeb2, ffn2, (int)MT, D, DFF, 0, stream);
    ln_kernel<<<dim3((int)MT), 256, 0, stream>>>(tmp1, ffn2, neg, neb, h, D);
  }

  // --- hoisted cross-attention K/V (identical math, computed once) ---
  for (int l = 0; l < Lc; ++l) {
    launch_gemm(h, decWk + (size_t)l * D * D, decbk + (size_t)l * D,
                crossK + (size_t)l * MT * D, (int)MT, D, D, 0, stream);
    launch_gemm(h, decWv + (size_t)l * D * D, decbv + (size_t)l * D,
                crossV + (size_t)l * MT * D, (int)MT, D, D, 0, stream);
  }

  // --- autoregressive decoder ---
  for (int s = 0; s < NS; ++s) {
    const int Tq = s + 1;
    const int Mq = Bc * Tq;
    add_pe<<<dim3((Mq * D + 255) / 256), 256, 0, stream>>>(dec, pe, dxa, Bc, Tq, 25, D);
    const dim3 fgrid(Bc * Hh, (Tq + 16 * FQW - 1) / (16 * FQW));
    for (int l = 0; l < Lc; ++l) {
      const float* Wq = decWq + (size_t)l * D * D; const float* bq = decbq + (size_t)l * D;
      const float* Wk = decWk + (size_t)l * D * D; const float* bk = decbk + (size_t)l * D;
      const float* Wv = decWv + (size_t)l * D * D; const float* bv = decbv + (size_t)l * D;
      const float* Wo = decWo + (size_t)l * D * D; const float* bo = decbo + (size_t)l * D;

      // self-attention (no mask in reference)
      launch_gemm(dxa, Wq, bq, dq, Mq, D, D, 0, stream);
      launch_gemm(dxa, Wk, bk, dk, Mq, D, D, 0, stream);
      launch_gemm(dxa, Wv, bv, dv, Mq, D, D, 0, stream);
      flash_wmma<<<fgrid, 32 * FQW, 0, stream>>>(dq, dk, dv, dattn, Tq, Tq);
      launch_gemm(dattn, Wo, bo, dt1, Mq, D, D, 0, stream);
      // cross-attention against hoisted K/V
      launch_gemm(dt1, Wq, bq, dq, Mq, D, D, 0, stream);
      flash_wmma<<<fgrid, 32 * FQW, 0, stream>>>(dq, crossK + (size_t)l * MT * D,
                                                 crossV + (size_t)l * MT * D, dattn, Tq, T);
      launch_gemm(dattn, Wo, bo, dt2, Mq, D, D, 0, stream);
      // FFN + residual LN
      launch_gemm(dt2, ffdW1, ffdb1, dffn1, Mq, DFF, D, 1, stream);
      launch_gemm(dffn1, ffdW2, ffdb2, dffn2, Mq, D, DFF, 0, stream);
      ln_kernel<<<dim3(Mq), 256, 0, stream>>>(dt2, dffn2, ndg, ndb, dxa, D);
    }
    step_out<<<dim3(Bc), 256, 0, stream>>>(dxa, Tq, outW, outb, tembW, tembb,
                                           preds, dec, s);
  }
}
